// GBST_87436944212641
// MI455X (gfx1250) — compile-verified, matrix operand order checked by reading, not run on hardware
//
#include <hip/hip_runtime.h>
#include <hip/hip_bf16.h>
#include <stdint.h>

typedef __attribute__((ext_vector_type(16))) _Float16 v16h;
typedef __attribute__((ext_vector_type(8)))  float    v8f;

#define B_    4
#define L_    4096
#define E_    512
#define V_    256
#define LC    192      // positions per workgroup (multiple of lcm(2,3,4)=12)
#define EC    128      // embedding columns per workgroup
#define HALO  8        // left halo (need up to 4; 8 keeps indices simple)
#define SEQN  (LC + 2*HALO)

// ---------------------------------------------------------------------------
// Kernel 1: tok_score[v] = dot(emb[v,:], score_w)  via V_WMMA_F32_16X16X32_F16
// One wave per 16-token tile (16 waves). B-matrix has score_w in column 0 only.
// Branch-free B construction: unconditional loads * lane-select constant.
// ---------------------------------------------------------------------------
__global__ __launch_bounds__(512) void tok_score_wmma(const float* __restrict__ emb,
                                                      const float* __restrict__ sw,
                                                      float* __restrict__ tok) {
  const int lane = threadIdx.x & 31;
  const int wave = threadIdx.x >> 5;   // 16 waves -> rows [wave*16, wave*16+16)
  const int mn   = lane & 15;
  const int half = lane >> 4;
  const int row  = wave * 16 + mn;
  const float sel = (mn == 0) ? 1.f : 0.f;   // only column N=0 carries score_w
  const float* erow = emb + (size_t)row * E_;

  v8f acc = {0.f, 0.f, 0.f, 0.f, 0.f, 0.f, 0.f, 0.f};
  for (int k0 = 0; k0 < E_; k0 += 32) {
    v16h a, b;
#pragma unroll
    for (int v = 0; v < 8; ++v) {
      // 16-bit A 16x32 layout: VGPR v<4 -> K = 2v + 8*half; v>=4 -> K = 16+2(v-4)+8*half
      const int kb = k0 + ((v < 4) ? (2 * v) : (16 + 2 * (v - 4))) + 8 * half;
      a[2 * v]     = (_Float16)erow[kb];
      a[2 * v + 1] = (_Float16)erow[kb + 1];
      b[2 * v]     = (_Float16)(sw[kb]     * sel);
      b[2 * v + 1] = (_Float16)(sw[kb + 1] * sel);
    }
    acc = __builtin_amdgcn_wmma_f32_16x16x32_f16(false, a, false, b,
                                                 (short)0, acc, false, false);
  }
  // C column N=0 lives in lane 0 (M = v) and lane 16 (M = v + 8)
  if (mn == 0) {
#pragma unroll
    for (int v = 0; v < 8; ++v) tok[wave * 16 + half * 8 + v] = acc[v];
  }
}

// ---------------------------------------------------------------------------
// CDNA5 async global->LDS copy (ASYNCcnt path, ISA ch.10 / 15.18.3)
// ---------------------------------------------------------------------------
__device__ __forceinline__ uint32_t lds_off(const void* p) {
  return (uint32_t)(uintptr_t)p;   // low 32 bits of generic LDS ptr = LDS offset
}
__device__ __forceinline__ void async_g2l_b128(uint32_t ldsaddr, uint64_t gaddr) {
  asm volatile("global_load_async_to_lds_b128 %0, %1, off"
               :: "v"(ldsaddr), "v"(gaddr) : "memory");
}
__device__ __forceinline__ void wait_async_zero() {
  asm volatile("s_wait_asynccnt 0" ::: "memory");
}

// ---------------------------------------------------------------------------
// Kernel 2: fused main kernel. Grid: (ceil(L/LC), E/EC, B), 256 threads.
// Phase B streams each source row ONCE per 12-position period, maintaining
// running prev-block means in registers (minimal LDS read traffic).
// ---------------------------------------------------------------------------
__global__ __launch_bounds__(256) void gbst_main(const int* __restrict__ seq,
                                                 const float* __restrict__ emb,
                                                 const float* __restrict__ tok,
                                                 float* __restrict__ out) {
  __shared__ __align__(16) float s_emb[V_ * EC];   // 128 KB: emb[:, e0:e0+EC]
  __shared__ float s_tok[V_];                      // 1 KB
  __shared__ int   s_seq[SEQN];                    // seq chunk + halo
  __shared__ __align__(16) float s_w[LC * 4];      // softmax weights per position

  const int t  = threadIdx.x;
  const int b  = blockIdx.z;
  const int e0 = blockIdx.y * EC;
  const int l0 = blockIdx.x * LC;
  const int nl = min(LC, L_ - l0);

  // ---- (1) async staging of the emb tile (32 x b128 per thread) -------------
  {
    const uint32_t lb = lds_off(&s_emb[0]);
#pragma unroll 4
    for (int i = 0; i < 32; ++i) {
      const int idx = i * 256 + t;
      const int row = idx >> 5;          // token row 0..255
      const int c4  = idx & 31;          // float4 column within EC
      async_g2l_b128(lb + (uint32_t)((row * EC + c4 * 4) * 4),
                     (uint64_t)(uintptr_t)(emb + row * E_ + e0 + c4 * 4));
    }
  }

  // ---- (2) regular staging of seq(+halo) and tok_score ----------------------
  if (t < SEQN) {
    const int p = l0 - HALO + t;
    s_seq[t] = (p >= 0 && p < L_) ? seq[b * L_ + p] : 0;  // 0 -> emb[PAD]=0
  }
  s_tok[t] = tok[t];                     // t covers exactly V_=256
  __syncthreads();

  // ---- (3) phase A: per-position softmax weights ----------------------------
  if (t < nl) {
    const int l = l0 + t;
    const float s0 = s_tok[s_seq[t + HALO]];
    float sc[3];
#pragma unroll
    for (int k = 0; k < 3; ++k) {
      const int bs = k + 2;
      const int g  = l / bs;
      if (g > 0) {                        // logit = mean tok over PREVIOUS bs-block
        const int ps = (g - 1) * bs;
        float sum = 0.f;
        for (int i = 0; i < bs; ++i) sum += s_tok[s_seq[ps - l0 + HALO + i]];
        sc[k] = sum / (float)bs;
      } else {                            // zero rep -> logit 0 (score_b cancels)
        sc[k] = 0.f;
      }
    }
    const float mx = fmaxf(fmaxf(s0, sc[0]), fmaxf(sc[1], sc[2]));
    float w0 = __expf(s0    - mx);
    float w1 = __expf(sc[0] - mx);
    float w2 = __expf(sc[1] - mx);
    float w3 = __expf(sc[2] - mx);
    const float inv = 1.f / (w0 + w1 + w2 + w3);
    s_w[t * 4 + 0] = w0 * inv;
    s_w[t * 4 + 1] = w1 * inv;
    s_w[t * 4 + 2] = w2 * inv;
    s_w[t * 4 + 3] = w3 * inv;
  }
  __syncthreads();

  // ---- (4) finish async staging --------------------------------------------
  wait_async_zero();                     // each wave drains its own ASYNCcnt
  __syncthreads();

  // ---- (5) phase B: stream one 12-position period per thread ----------------
  // For period start p0 (chunk-relative, mult of 12), iterate source rows
  // jr = p0-4 .. p0+11. At block boundaries (compile-time pattern since
  // p0 % 12 == 0) snapshot prev-block means Mk = Ak/bs. Each row read once.
  const int nper  = (nl + 11) / 12;      // 16 when chunk is full
  const int total = nper * 32;           // (period, e4) pairs
  for (int idx = t; idx < total; idx += 256) {
    const int e4  = idx & 31;
    const int per = idx >> 5;            // uniform within a wave
    const int p0  = per * 12;

    float M2x=0,M2y=0,M2z=0,M2w=0, M3x=0,M3y=0,M3z=0,M3w=0, M4x=0,M4y=0,M4z=0,M4w=0;
    float A2x=0,A2y=0,A2z=0,A2w=0, A3x=0,A3y=0,A3z=0,A3w=0, A4x=0,A4y=0,A4z=0,A4w=0;

#pragma unroll
    for (int i = 0; i < 16; ++i) {
      const int jr = p0 + i - 4;         // chunk-relative source position
      // block-boundary snapshots: jr%4==0 <=> i%4==0; jr%3==0 <=> i%3==1; jr%2==0 <=> i%2==0
      if ((i & 3) == 0) { M4x=A4x*0.25f; M4y=A4y*0.25f; M4z=A4z*0.25f; M4w=A4w*0.25f;
                          A4x=0; A4y=0; A4z=0; A4w=0; }
      if ((i % 3) == 1) { const float r3 = 1.f/3.f;
                          M3x=A3x*r3; M3y=A3y*r3; M3z=A3z*r3; M3w=A3w*r3;
                          A3x=0; A3y=0; A3z=0; A3w=0; }
      if ((i & 1) == 0) { M2x=A2x*0.5f; M2y=A2y*0.5f; M2z=A2z*0.5f; M2w=A2w*0.5f;
                          A2x=0; A2y=0; A2z=0; A2w=0; }

      float rx = 0.f, ry = 0.f, rz = 0.f, rw = 0.f;
      if (jr < nl && (l0 + jr) >= 0) {   // uniform-per-wave guard
        const int tokv = s_seq[jr + HALO];
        const float4 r = *(const float4*)&s_emb[tokv * EC + e4 * 4];
        rx = r.x; ry = r.y; rz = r.z; rw = r.w;
      }

      if (i >= 4 && jr < nl) {           // emit output for position jr
        const float4 w = *(const float4*)&s_w[jr * 4];   // broadcast in wave
        float ox = w.x * rx; ox = fmaf(w.y, M2x, ox); ox = fmaf(w.z, M3x, ox); ox = fmaf(w.w, M4x, ox);
        float oy = w.x * ry; oy = fmaf(w.y, M2y, oy); oy = fmaf(w.z, M3y, oy); oy = fmaf(w.w, M4y, oy);
        float oz = w.x * rz; oz = fmaf(w.y, M2z, oz); oz = fmaf(w.z, M3z, oz); oz = fmaf(w.w, M4z, oz);
        float ow = w.x * rw; ow = fmaf(w.y, M2w, ow); ow = fmaf(w.z, M3w, ow); ow = fmaf(w.w, M4w, ow);
        float4 o; o.x = ox; o.y = oy; o.z = oz; o.w = ow;
        *(float4*)&out[(((size_t)b * L_ + (l0 + jr)) * E_) + e0 + e4 * 4] = o;
      }

      A2x += rx; A2y += ry; A2z += rz; A2w += rw;
      A3x += rx; A3y += ry; A3z += rz; A3w += rw;
      A4x += rx; A4y += ry; A4z += rz; A4w += rw;
    }
  }
}

// ---------------------------------------------------------------------------
extern "C" void kernel_launch(void* const* d_in, const int* in_sizes, int n_in,
                              void* d_out, int out_size, void* d_ws, size_t ws_size,
                              hipStream_t stream) {
  (void)in_sizes; (void)n_in; (void)out_size; (void)ws_size;
  const int*   seq = (const int*)d_in[0];
  // d_in[1] = group_id: implied by pos//bs, unused
  const float* emb = (const float*)d_in[2];
  const float* sw  = (const float*)d_in[3];
  // d_in[4] = score_b: common additive logit, cancels in softmax
  float* out = (float*)d_out;
  float* tok = (float*)d_ws;             // 256 floats of scratch

  tok_score_wmma<<<1, 512, 0, stream>>>(emb, sw, tok);

  dim3 grid((L_ + LC - 1) / LC, E_ / EC, B_);
  gbst_main<<<grid, 256, 0, stream>>>(seq, emb, tok, out);
}
